// ParticleNetTagger_46222438039812
// MI455X (gfx1250) — matrix operand order repaired and checked by reading, not compiled, not run on hardware
//
#include <hip/hip_runtime.h>
#include <hip/hip_bf16.h>

typedef __attribute__((ext_vector_type(16))) _Float16 v16h;
typedef __attribute__((ext_vector_type(8)))  float    v8f;

// ---------------------------------------------------------------------------
// Input pointer table (setup_inputs() dict order, params flattened in
// insertion order; ec0.sc is None -> no leaf).
// ---------------------------------------------------------------------------
enum {
  I_PF_POINTS = 0, I_CPF = 1, I_PF_MASK = 2, I_SV_POINTS = 3, I_VTX = 4, I_SV_MASK = 5,
  I_PF_IN_G = 6,   // 6..9  : pf_conv.bn_in  g,b,m,v (30)
  I_PF_W = 10,     //       : pf_conv.w (32x30)
  I_PF_OUT_G = 11, // 11..14: pf_conv.bn_out g,b,m,v (32)
  I_SV_IN_G = 15,  // 15..18: sv_conv.bn_in (15)
  I_SV_W = 19,     //       : sv_conv.w (32x15)
  I_SV_OUT_G = 20, // 20..23: sv_conv.bn_out (32)
  I_BNFTS_G = 24,  // 24..27: bn_fts (32)
  I_EC0_W0 = 28, I_EC0_BN0 = 29,   // w(32x64), bn(32) g,b,m,v
  I_EC0_W1 = 33, I_EC0_BN1 = 34,   // w(32x32)
  I_EC0_W2 = 38, I_EC0_BN2 = 39,   // w(32x32)
  I_EC1_W0 = 43, I_EC1_BN0 = 44,   // w(64x64), bn(64)
  I_EC1_W1 = 48, I_EC1_BN1 = 49,
  I_EC1_W2 = 53, I_EC1_BN2 = 54,
  I_EC1_SCW = 58, I_EC1_SCBN = 59, // sc w(64x32), bn(64)
  I_FUS_W = 63, I_FUS_BN = 64,     // w(128x96), bn(128)
  I_FC1_W = 68, I_FC1_B = 69,
  I_FC2_W = 70, I_FC2_B = 71,
  N_PTRS = 72
};

struct Ptrs { const float* p[N_PTRS]; };

#define NPF 50
#define NSV 5
#define NPT 55     // total points
#define NPAD 64    // padded column stride for LDS matrices
#define KNN 7
#define KPAD 8
#define NTILES 28  // 56 points * 8 k-slots / 16 columns

// ---------------------------------------------------------------------------
// WMMA helpers (16x16x32 f16 -> f32)
// ---------------------------------------------------------------------------
__device__ __forceinline__ v8f wmma16(v16h a, v16h b, v8f c) {
  return __builtin_amdgcn_wmma_f32_16x16x32_f16(
      /*neg_a=*/false, a, /*neg_b=*/false, b,
      /*c_mod=*/(short)0, c, /*reuse_a=*/false, /*reuse_b=*/false);
}

// A fragment: 16(M) x 32(K) f16, W row-major [O][C] in LDS.
// lanes 0-15 -> K base 0, lanes 16-31 -> K base 16;
// element h : Kb+h (contiguous, matches B fragments).
__device__ __forceinline__ v16h load_A_frag(const _Float16* W, int ldw,
                                            int mt, int kt, int lane) {
  const int M  = mt * 16 + (lane & 15);
  const int kb = (lane < 16) ? 0 : 16;
  v16h a;
#pragma unroll
  for (int h = 0; h < 16; ++h) {
    const int j = h >> 1, pb = h & 1;
    const int K = kb + 2 * j + pb;
    a[h] = W[M * ldw + kt * 32 + K];
  }
  return a;
}

// B fragment: 32(K) x 16(N) f16 from per-wave activation tile [C][16].
// Lanes 0-15 hold K=0..15, lanes 16-31 hold K=16..31 (contiguous, matches A fragments).
__device__ __forceinline__ v16h load_B_act(const _Float16* act, int kt, int lane) {
  const int N  = lane & 15;
  const int kb = (lane < 16) ? 0 : 16;
  v16h b;
#pragma unroll
  for (int h = 0; h < 16; ++h) b[h] = act[(kt * 32 + kb + h) * 16 + N];
  return b;
}

// Apply BN+ReLU to D fragments and store to per-wave act tile as f16.
template <int MT>
__device__ __forceinline__ void store_act(const v8f* d, const float* S,
                                          const float* B, _Float16* act,
                                          int N, int moD) {
#pragma unroll
  for (int mt = 0; mt < MT; ++mt) {
#pragma unroll
    for (int r = 0; r < 8; ++r) {
      const int o = mt * 16 + moD + r;
      float v = fmaf(d[mt][r], S[o], B[o]);
      act[o * 16 + N] = (_Float16)fmaxf(v, 0.f);
    }
  }
}

// ---------------------------------------------------------------------------
// One EdgeConv conv-stack (3 layers) via WMMA. COUT = 32 (EC1) or 64 (EC2).
// ftsT  : LDS f32 [64][32] input features TRANSPOSED (rows >= NPT zeroed)
// idx   : LDS int [56][8] neighbor indices (row 55 = 55, slot 7 = self)
// outAcc: LDS f32 [COUT][NPAD], pre-zeroed; receives mean over k
// w1/w2/w3: LDS f16 weights (row-major [O][C]); bnS/bnB: [3][64] scale/bias
// act   : per-wave LDS f16 [64*16] staging tile
// ---------------------------------------------------------------------------
template <int COUT>
__device__ void edge_conv_wmma(const float* ftsT, const int* idx,
                               const _Float16* w1, const _Float16* w2,
                               const _Float16* w3, const float* bnS,
                               const float* bnB, float* outAcc,
                               _Float16* act, int lane, int wave) {
  constexpr int MT  = COUT / 16;
  constexpr int KT2 = COUT / 32;
  const int N   = lane & 15;
  const int kb  = (lane < 16) ? 0 : 16;  // B-fragment K base
  const int moD = (lane < 16) ? 0 : 8;   // D-fragment M offset
  const v8f zero8 = {0.f, 0.f, 0.f, 0.f, 0.f, 0.f, 0.f, 0.f};

  for (int ct = wave; ct < NTILES; ct += 8) {
    const int colg = ct * 16 + N;      // global (point,k) column
    const int p    = colg >> 3;        // point index (0..55)
    const int kk   = colg & 7;         // k slot (7 = padding)
    const int nbr  = idx[p * KPAD + kk];   // always valid (padded rows)
    const float wk = (p < NPT && kk < KNN) ? (1.0f / 7.0f) : 0.f;

    v8f d[MT];
#pragma unroll
    for (int mt = 0; mt < MT; ++mt) d[mt] = zero8;

    // ---- layer 1: K=64 graph feature; kt=0 -> all center, kt=1 -> all diff
    {
      const float* ctr = ftsT + p * 32 + kb;
      const float* nbp = ftsT + nbr * 32 + kb;
      v16h bf0, bf1;
#pragma unroll
      for (int h = 0; h < 16; ++h) {
        const float cv = ctr[h];
        bf0[h] = (_Float16)cv;
        bf1[h] = (_Float16)(nbp[h] - cv);
      }
#pragma unroll
      for (int mt = 0; mt < MT; ++mt)
        d[mt] = wmma16(load_A_frag(w1, 64, mt, 0, lane), bf0, d[mt]);
#pragma unroll
      for (int mt = 0; mt < MT; ++mt)
        d[mt] = wmma16(load_A_frag(w1, 64, mt, 1, lane), bf1, d[mt]);
    }
    store_act<MT>(d, bnS, bnB, act, N, moD);
    __builtin_amdgcn_wave_barrier();

    // ---- layer 2: K = COUT ----
#pragma unroll
    for (int mt = 0; mt < MT; ++mt) d[mt] = zero8;
#pragma unroll
    for (int kt = 0; kt < KT2; ++kt) {
      v16h bf = load_B_act(act, kt, lane);
#pragma unroll
      for (int mt = 0; mt < MT; ++mt)
        d[mt] = wmma16(load_A_frag(w2, COUT, mt, kt, lane), bf, d[mt]);
    }
    __builtin_amdgcn_wave_barrier();
    store_act<MT>(d, bnS + 64, bnB + 64, act, N, moD);
    __builtin_amdgcn_wave_barrier();

    // ---- layer 3: K = COUT ----
#pragma unroll
    for (int mt = 0; mt < MT; ++mt) d[mt] = zero8;
#pragma unroll
    for (int kt = 0; kt < KT2; ++kt) {
      v16h bf = load_B_act(act, kt, lane);
#pragma unroll
      for (int mt = 0; mt < MT; ++mt)
        d[mt] = wmma16(load_A_frag(w3, COUT, mt, kt, lane), bf, d[mt]);
    }
    // BN+ReLU then mean over k (padded cols contribute 0 via wk)
#pragma unroll
    for (int mt = 0; mt < MT; ++mt) {
#pragma unroll
      for (int r = 0; r < 8; ++r) {
        const int o = mt * 16 + moD + r;
        float v = fmaf(d[mt][r], bnS[128 + o], bnB[128 + o]);
        atomicAdd(&outAcc[o * NPAD + p], fmaxf(v, 0.f) * wk);
      }
    }
    __builtin_amdgcn_wave_barrier();
  }
}

// ---------------------------------------------------------------------------
__device__ __forceinline__ void bn_prep(const float* g, const float* bb,
                                        const float* m, const float* v, int C,
                                        float* S, float* B, int tid) {
  if (tid < C) {
    float s = g[tid] * rsqrtf(v[tid] + 1e-5f);
    S[tid] = s;
    B[tid] = bb[tid] - m[tid] * s;
  }
}

// kNN: top-(k+1) of neg squared distance, drop self-top, jax tie-break (index asc).
__device__ void knn_build(const float* pts, int D, int* idxOut, int tid) {
  if (tid < NPT) {
    float bv[8];
    int   bi[8];
#pragma unroll
    for (int j = 0; j < 8; ++j) { bv[j] = -3.4e38f; bi[j] = 0; }
    for (int m = 0; m < NPT; ++m) {
      float dist = 0.f;
      for (int c = 0; c < D; ++c) {
        float dd = pts[c * NPAD + tid] - pts[c * NPAD + m];
        dist = fmaf(dd, dd, dist);
      }
      float v = -dist;
      if (v > bv[7]) {
        bv[7] = v; bi[7] = m;
#pragma unroll
        for (int j = 7; j > 0; --j) {
          if (bv[j] > bv[j - 1]) {
            float tv = bv[j]; bv[j] = bv[j - 1]; bv[j - 1] = tv;
            int   ti = bi[j]; bi[j] = bi[j - 1]; bi[j - 1] = ti;
          }
        }
      }
    }
#pragma unroll
    for (int kk = 0; kk < KNN; ++kk) idxOut[tid * KPAD + kk] = bi[kk + 1];
    idxOut[tid * KPAD + 7] = tid;          // padded k slot -> self (diff = 0)
  }
  if (tid >= 56 && tid < 64) idxOut[NPT * KPAD + (tid - 56)] = NPT;  // pad row
}

// ---------------------------------------------------------------------------
// Fused ParticleNet forward: one workgroup (256 threads = 8 waves) per jet.
// ---------------------------------------------------------------------------
__global__ void __launch_bounds__(256)
particlenet_fused_kernel(Ptrs P, float* __restrict__ out) {
  __shared__ float sPts [32 * NPAD];   // knn coords (then EC2-sc weight stage)
  __shared__ float sFts [32 * NPAD];   // bn_fts output [c][n]
  __shared__ float sFtsT[NPAD * 32];   // transposed features [n][c] (f32)
  __shared__ float sOut1[32 * NPAD];   // EC1 mean-acc then EC1 output
  __shared__ float sOut2[64 * NPAD];   // input stage / EC2 mean-acc / EC2 out
  __shared__ float sBnS[4 * 64], sBnB[4 * 64];
  __shared__ float sFtsS[32], sFtsB[32];
  __shared__ float sMask[NPAD];
  __shared__ float sPool[128], sFc1[128];
  __shared__ float sCnt;
  __shared__ int   sIdx[56 * KPAD];
  __shared__ _Float16 sWbuf[3 * 4096];   // EC weights w1/w2/w3; then fusion W
  __shared__ _Float16 sAct[8][64 * 16];  // per-wave staging; then catT[64][96]

  const int b    = blockIdx.x;
  const int tid  = threadIdx.x;
  const int lane = tid & 31;
  const int wave = tid >> 5;
  _Float16* catT = &sAct[0][0];

  // ---------------- phase 1: BN prep (feature convs) + mask/points --------
  bn_prep(P.p[I_PF_IN_G],  P.p[I_PF_IN_G + 1],  P.p[I_PF_IN_G + 2],  P.p[I_PF_IN_G + 3],  30, sBnS,       sBnB,       tid);
  bn_prep(P.p[I_PF_OUT_G], P.p[I_PF_OUT_G + 1], P.p[I_PF_OUT_G + 2], P.p[I_PF_OUT_G + 3], 32, sBnS + 64,  sBnB + 64,  tid);
  bn_prep(P.p[I_SV_IN_G],  P.p[I_SV_IN_G + 1],  P.p[I_SV_IN_G + 2],  P.p[I_SV_IN_G + 3],  15, sBnS + 128, sBnB + 128, tid);
  bn_prep(P.p[I_SV_OUT_G], P.p[I_SV_OUT_G + 1], P.p[I_SV_OUT_G + 2], P.p[I_SV_OUT_G + 3], 32, sBnS + 192, sBnB + 192, tid);
  bn_prep(P.p[I_BNFTS_G],  P.p[I_BNFTS_G + 1],  P.p[I_BNFTS_G + 2],  P.p[I_BNFTS_G + 3],  32, sFtsS,      sFtsB,      tid);
  if (tid < NPAD) {
    float mk = 0.f, p0 = 0.f, p1 = 0.f;
    if (tid < NPF) {
      mk = P.p[I_PF_MASK][b * NPF + tid];
      p0 = P.p[I_PF_POINTS][(b * 2 + 0) * NPF + tid];
      p1 = P.p[I_PF_POINTS][(b * 2 + 1) * NPF + tid];
    } else if (tid < NPT) {
      int j = tid - NPF;
      mk = P.p[I_SV_MASK][b * NSV + j];
      p0 = P.p[I_SV_POINTS][(b * 2 + 0) * NSV + j];
      p1 = P.p[I_SV_POINTS][(b * 2 + 1) * NSV + j];
    }
    float shift = (mk == 0.f) ? 1e9f : 0.f;
    sMask[tid] = mk;
    sPts[0 * NPAD + tid] = p0 * mk + shift;
    sPts[1 * NPAD + tid] = p1 * mk + shift;
  }
  __syncthreads();

  // ---------------- phase 1b: stage bn_in(x*mask) into sOut2 scratch ------
  for (int it = tid; it < 30 * NPF; it += 256) {
    const int c = it / NPF, n = it % NPF;
    float x = P.p[I_CPF][(b * 30 + c) * NPF + n] * sMask[n];
    sOut2[c * NPAD + n] = fmaf(x, sBnS[c], sBnB[c]);
  }
  for (int it = tid; it < 15 * NSV; it += 256) {
    const int c = it / NSV, j = it % NSV;
    float x = P.p[I_VTX][(b * 15 + c) * NSV + j] * sMask[NPF + j];
    sOut2[c * NPAD + NPF + j] = fmaf(x, sBnS[128 + c], sBnB[128 + c]);
  }
  if (tid == 0) {
    float s = 0.f;
    for (int n = 0; n < NPT; ++n) s += sMask[n];
    sCnt = fmaxf(s, 1.f);
  }
  __syncthreads();

  // ---------------- phase 2: feature convs + bn_fts -----------------------
  for (int it = tid; it < 32 * NPT; it += 256) {
    const int o = it & 31, n = it >> 5;
    const float mk = sMask[n];
    float acc = 0.f;
    if (n < NPF) {
      for (int c = 0; c < 30; ++c)
        acc = fmaf(P.p[I_PF_W][o * 30 + c], sOut2[c * NPAD + n], acc);
      acc = fmaf(acc, sBnS[64 + o], sBnB[64 + o]);
    } else {
      for (int c = 0; c < 15; ++c)
        acc = fmaf(P.p[I_SV_W][o * 15 + c], sOut2[c * NPAD + n], acc);
      acc = fmaf(acc, sBnS[192 + o], sBnB[192 + o]);
    }
    acc = fmaxf(acc, 0.f) * mk;                 // feature conv output
    acc = fmaf(acc, sFtsS[o], sFtsB[o]) * mk;   // bn_fts
    sFts[o * NPAD + n] = acc;
    sFtsT[n * 32 + o]  = acc;
  }
  for (int it = tid; it < 9 * 32; it += 256)    // zero pad rows 55..63
    sFtsT[(NPT + it / 32) * 32 + (it & 31)] = 0.f;
  __syncthreads();

  // ---------------- phase 3: kNN(2d) + EC1 weights/BN ---------------------
  knn_build(sPts, 2, sIdx, tid);
  for (int it = tid; it < 32 * 64; it += 256) sWbuf[it] = (_Float16)P.p[I_EC0_W0][it];
  for (int it = tid; it < 32 * 32; it += 256) {
    sWbuf[4096 + it] = (_Float16)P.p[I_EC0_W1][it];
    sWbuf[8192 + it] = (_Float16)P.p[I_EC0_W2][it];
  }
  bn_prep(P.p[I_EC0_BN0], P.p[I_EC0_BN0 + 1], P.p[I_EC0_BN0 + 2], P.p[I_EC0_BN0 + 3], 32, sBnS,       sBnB,       tid);
  bn_prep(P.p[I_EC0_BN1], P.p[I_EC0_BN1 + 1], P.p[I_EC0_BN1 + 2], P.p[I_EC0_BN1 + 3], 32, sBnS + 64,  sBnB + 64,  tid);
  bn_prep(P.p[I_EC0_BN2], P.p[I_EC0_BN2 + 1], P.p[I_EC0_BN2 + 2], P.p[I_EC0_BN2 + 3], 32, sBnS + 128, sBnB + 128, tid);
  for (int it = tid; it < 32 * NPAD; it += 256) sOut1[it] = 0.f;
  __syncthreads();

  // ---------------- phase 4: EdgeConv1 (WMMA, COUT=32) --------------------
  edge_conv_wmma<32>(sFtsT, sIdx, sWbuf, sWbuf + 4096, sWbuf + 8192,
                     sBnS, sBnB, sOut1, &sAct[wave][0], lane, wave);
  __syncthreads();

  // ---------------- phase 5: EC1 shortcut + EC2 coords --------------------
  for (int it = tid; it < 32 * NPT; it += 256) {
    const int o = it & 31, n = it >> 5;
    float v = fmaxf(sFts[o * NPAD + n] + sOut1[o * NPAD + n], 0.f) * sMask[n];
    sOut1[o * NPAD + n] = v;                                 // EC1 output
    sFtsT[n * 32 + o]   = v;                                 // EC2 input (T)
    sPts[o * NPAD + n]  = v + ((sMask[n] == 0.f) ? 1e9f : 0.f);
  }
  for (int it = tid; it < 64 * NPAD; it += 256) sOut2[it] = 0.f;
  __syncthreads();

  // ---------------- phase 6: kNN(32d) + EC2 weights/BN --------------------
  knn_build(sPts, 32, sIdx, tid);
  for (int it = tid; it < 64 * 64; it += 256) {
    sWbuf[it]        = (_Float16)P.p[I_EC1_W0][it];
    sWbuf[4096 + it] = (_Float16)P.p[I_EC1_W1][it];
    sWbuf[8192 + it] = (_Float16)P.p[I_EC1_W2][it];
  }
  bn_prep(P.p[I_EC1_BN0],  P.p[I_EC1_BN0 + 1],  P.p[I_EC1_BN0 + 2],  P.p[I_EC1_BN0 + 3],  64, sBnS,       sBnB,       tid);
  bn_prep(P.p[I_EC1_BN1],  P.p[I_EC1_BN1 + 1],  P.p[I_EC1_BN1 + 2],  P.p[I_EC1_BN1 + 3],  64, sBnS + 64,  sBnB + 64,  tid);
  bn_prep(P.p[I_EC1_BN2],  P.p[I_EC1_BN2 + 1],  P.p[I_EC1_BN2 + 2],  P.p[I_EC1_BN2 + 3],  64, sBnS + 128, sBnB + 128, tid);
  bn_prep(P.p[I_EC1_SCBN], P.p[I_EC1_SCBN + 1], P.p[I_EC1_SCBN + 2], P.p[I_EC1_SCBN + 3], 64, sBnS + 192, sBnB + 192, tid);
  __syncthreads();

  // ---------------- phase 7: EdgeConv2 (WMMA, COUT=64) --------------------
  edge_conv_wmma<64>(sFtsT, sIdx, sWbuf, sWbuf + 4096, sWbuf + 8192,
                     sBnS, sBnB, sOut2, &sAct[wave][0], lane, wave);
  __syncthreads();

  // ---------------- phase 8a: stage fusion/sc weights, catT out1 half -----
  bn_prep(P.p[I_FUS_BN], P.p[I_FUS_BN + 1], P.p[I_FUS_BN + 2], P.p[I_FUS_BN + 3], 128, sBnS, sBnB, tid);
  if (tid < 128) sPool[tid] = 0.f;
  for (int it = tid; it < 64 * 32; it += 256) sPts[it] = P.p[I_EC1_SCW][it];
  for (int it = tid; it < 128 * 96; it += 256) sWbuf[it] = (_Float16)P.p[I_FUS_W][it];
  for (int it = tid; it < NPAD * 32; it += 256) {
    const int n = it >> 5, c = it & 31;
    catT[n * 96 + c] = (_Float16)sFtsT[n * 32 + c];   // out1 (rows>=55 are 0)
  }
  for (int it = tid; it < 9 * 64; it += 256)          // zero out2 pad rows
    catT[(NPT + it / 64) * 96 + 32 + (it & 63)] = (_Float16)0.f;
  __syncthreads();

  // ---------------- phase 8b: EC2 shortcut + catT out2 half ---------------
  for (int it = tid; it < 64 * NPT; it += 256) {
    const int o = it & 63, n = it >> 6;
    float sc = 0.f;
    for (int c = 0; c < 32; ++c)
      sc = fmaf(sPts[o * 32 + c], sOut1[c * NPAD + n], sc);
    sc = fmaf(sc, sBnS[192 + o], sBnB[192 + o]);
    float v = fmaxf(sc + sOut2[o * NPAD + n], 0.f) * sMask[n]; // EC2 output
    sOut2[o * NPAD + n] = v;
    catT[n * 96 + 32 + o] = (_Float16)v;
  }
  __syncthreads();

  // ---------------- phase 9: fusion conv via WMMA + masked pool -----------
  if (wave < 4) {
    const int N   = lane & 15;
    const int kb  = (lane < 16) ? 0 : 16;
    const int moD = (lane < 16) ? 0 : 8;
    const int n   = wave * 16 + N;
    const float mk = sMask[n];
    const v8f zero8 = {0.f, 0.f, 0.f, 0.f, 0.f, 0.f, 0.f, 0.f};
#pragma unroll
    for (int mt = 0; mt < 8; ++mt) {
      v8f d = zero8;
#pragma unroll
      for (int kt = 0; kt < 3; ++kt) {
        v16h bf;
#pragma unroll
        for (int h = 0; h < 16; ++h) bf[h] = catT[n * 96 + kt * 32 + kb + h];
        d = wmma16(load_A_frag(sWbuf, 96, mt, kt, lane), bf, d);
      }
#pragma unroll
      for (int r = 0; r < 8; ++r) {
        const int o = mt * 16 + moD + r;
        float v = fmaf(d[r], sBnS[o], sBnB[o]);
        atomicAdd(&sPool[o], fmaxf(v, 0.f) * mk);
      }
    }
  }
  __syncthreads();
  if (tid < 128) sPool[tid] *= (1.0f / sCnt);
  __syncthreads();

  // ---------------- phase 10: FC head -------------------------------------
  if (tid < 128) {
    float acc = P.p[I_FC1_B][tid];
    for (int c = 0; c < 128; ++c)
      acc = fmaf(P.p[I_FC1_W][tid * 128 + c], sPool[c], acc);
    sFc1[tid] = fmaxf(acc, 0.f);
  }
  __syncthreads();
  if (tid < 4) {
    float acc = P.p[I_FC2_B][tid];
    for (int c = 0; c < 128; ++c)
      acc = fmaf(P.p[I_FC2_W][tid * 128 + c], sFc1[c], acc);
    out[b * 4 + tid] = acc;
  }
}

// ---------------------------------------------------------------------------
extern "C" void kernel_launch(void* const* d_in, const int* in_sizes, int n_in,
                              void* d_out, int out_size, void* d_ws, size_t ws_size,
                              hipStream_t stream) {
  Ptrs P;
  for (int i = 0; i < N_PTRS; ++i)
    P.p[i] = (i < n_in) ? (const float*)d_in[i] : (const float*)d_ws;
  const int B = in_sizes[0] / (2 * NPF);  // 4096
  hipLaunchKernelGGL(particlenet_fused_kernel, dim3(B), dim3(256), 0, stream,
                     P, (float*)d_out);
}
